// SelfAttnPropagation_48387101557136
// MI455X (gfx1250) — compile-verified, matrix operand order checked by reading, not run on hardware
//
#include <hip/hip_runtime.h>
#include <math.h>
#include <stdint.h>

#define BATCH 4
#define CH    128
#define HH    64
#define WW    96
#define HW    (HH * WW)            // 6144 tokens per batch
#define QT16  (HW / 16)            // 384 16-row key/query tiles per batch
#define TOTAL_TILES (BATCH * QT16) // 1536
#define PROJ_WPB 8
#define PROJ_BLOCKS (TOTAL_TILES / PROJ_WPB)   // 192
#define ATTN_WPB 16                             // 16 waves = 512 threads / block
#define ATTN_BLOCKS (TOTAL_TILES / ATTN_WPB)    // 96 (each block single-batch: 384%16==0)
#define KTILE_BYTES (16 * CH * 2)               // 4096 B per K tile

typedef __attribute__((ext_vector_type(16))) _Float16 v16h;
typedef __attribute__((ext_vector_type(8)))  _Float16 v8h;
typedef __attribute__((ext_vector_type(8)))  float    v8f;

// K-index pattern for the 16-bit 16x32 WMMA fragment layout (ISA 7.12.2):
// lanes 0-15 cover K={0..7,16..23}, lanes 16-31 cover K={8..15,24..31} per 32-chunk.
__device__ __forceinline__ int kpat(int kc, int khalf, int e) {
    return kc * 32 + (e >> 3) * 16 + khalf * 8 + (e & 7);
}

// Build one 16-half fragment from a contiguous f16 row (two b128 loads).
__device__ __forceinline__ v16h frag_from_row(const _Float16* row, int kc, int khalf) {
    v8h lo = *(const v8h*)(row + kc * 32 + khalf * 8);        // K = kc*32 + khalf*8 + 0..7
    v8h hi = *(const v8h*)(row + kc * 32 + 16 + khalf * 8);   // K = kc*32+16+khalf*8 + 0..7
    return __builtin_shufflevector(lo, hi, 0,1,2,3,4,5,6,7,8,9,10,11,12,13,14,15);
}

// ---------------- Stage 0: one-shot f32 -> f16 weight conversion ----------------
__global__ void __launch_bounds__(256)
cvt_weights_kernel(const float* __restrict__ qw, const float* __restrict__ kw,
                   _Float16* __restrict__ qwh, _Float16* __restrict__ kwh) {
    int i = blockIdx.x * 256 + threadIdx.x;      // 16384 elements each
    qwh[i] = (_Float16)qw[i];
    kwh[i] = (_Float16)kw[i];
}

// GEMM of one 16x128 A tile against W^T (f16 weights), +bias, store f16 row-major.
__device__ __forceinline__ void proj_tile(const v16h a[4],
                                          const _Float16* __restrict__ wh,
                                          const float* __restrict__ bias,
                                          _Float16* __restrict__ outbase,
                                          int m, int khalf) {
#pragma unroll
    for (int ot = 0; ot < 8; ++ot) {                         // 8 x 16 output columns
        const _Float16* wrow = wh + (size_t)(ot * 16 + m) * CH; // row o of W (y = x@W^T)
        v8f acc = {};
#pragma unroll
        for (int kc = 0; kc < 4; ++kc) {
            v16h bf = frag_from_row(wrow, kc, khalf);
            acc = __builtin_amdgcn_wmma_f32_16x16x32_f16(
                false, a[kc], false, bf, (short)0, acc, false, false);
        }
        float bb = bias[ot * 16 + m];
#pragma unroll
        for (int r = 0; r < 8; ++r) {
            // D layout: VGPR r -> row M=r (lanes 0-15) / M=r+8 (lanes 16-31), lane = col
            float v = acc[r] + bb;
            outbase[(size_t)(r + khalf * 8) * CH + ot * 16 + m] = (_Float16)v;
        }
    }
}

// Stage 1a: Q = tokens @ q_w^T + q_b ; tokens[n,c] = feature0[b,c,n] (strided gather)
__global__ void __launch_bounds__(256)
proj_q_kernel(const float* __restrict__ f0, const _Float16* __restrict__ qwh,
              const float* __restrict__ qb, _Float16* __restrict__ Q) {
    int wave = threadIdx.x >> 5, lane = threadIdx.x & 31;
    int tile = blockIdx.x * PROJ_WPB + wave;
    int b = tile / QT16, row0 = (tile % QT16) * 16;
    int m = lane & 15, khalf = lane >> 4;

    const float* src = f0 + (size_t)b * CH * HW + (row0 + m); // channel stride = HW
    v16h a[4];
#pragma unroll
    for (int kc = 0; kc < 4; ++kc)
#pragma unroll
        for (int e = 0; e < 16; ++e)
            a[kc][e] = (_Float16)src[(size_t)kpat(kc, khalf, e) * HW];

    proj_tile(a, qwh, qb, Q + ((size_t)b * HW + row0) * CH, m, khalf);
}

// Stage 1b: K = Q @ k_w^T + k_b ; Q is f16 row-major (serial dependency via stream order)
__global__ void __launch_bounds__(256)
proj_k_kernel(const _Float16* __restrict__ Q, const _Float16* __restrict__ kwh,
              const float* __restrict__ kb, _Float16* __restrict__ K) {
    int wave = threadIdx.x >> 5, lane = threadIdx.x & 31;
    int tile = blockIdx.x * PROJ_WPB + wave;
    int b = tile / QT16, row0 = (tile % QT16) * 16;
    int m = lane & 15, khalf = lane >> 4;

    const _Float16* src = Q + ((size_t)b * HW + row0 + m) * CH;
    v16h a[4];
#pragma unroll
    for (int kc = 0; kc < 4; ++kc)
        a[kc] = frag_from_row(src, kc, khalf);

    proj_tile(a, kwh, kb, K + ((size_t)b * HW + row0) * CH, m, khalf);
}

// Issue the async global->LDS copy of one 4KB K tile (first 256 threads, 16B each).
__device__ __forceinline__ void issue_ktile(const _Float16* kbase, int kt,
                                            _Float16* ldsbuf, int tid) {
    if (tid < 256) {   // waves 0..7 : wave-uniform branch
        uint64_t gaddr = (uint64_t)(uintptr_t)kbase
                       + (uint64_t)kt * KTILE_BYTES + (uint64_t)tid * 16;
        unsigned  laddr = (unsigned)(uintptr_t)ldsbuf + (unsigned)tid * 16;
        asm volatile("global_load_async_to_lds_b128 %0, %1, off"
                     :: "v"(laddr), "v"(gaddr) : "memory");
    }
}

// Stage 2: flash attention. 16 waves/block share each K tile via async-LDS double buffer.
__global__ void __launch_bounds__(512)
attn_kernel(const _Float16* __restrict__ Q, const _Float16* __restrict__ K,
            const float* __restrict__ flow, float* __restrict__ out) {
    __shared__ __align__(16) _Float16 ktile[2][16 * CH];    // 2 x 4KB double buffer

    int tid = threadIdx.x;
    int wave = tid >> 5, lane = tid & 31;
    int tile = blockIdx.x * ATTN_WPB + wave;
    int b = tile / QT16, row0 = (tile % QT16) * 16;         // whole block same batch
    int m = lane & 15, khalf = lane >> 4;
    const float scale = 0.0883883476483184f;                // 1/sqrt(128)

    // Q tile A-fragments held in registers for the whole key sweep
    const _Float16* qsrc = Q + ((size_t)b * HW + row0 + m) * CH;
    v16h aq[4];
#pragma unroll
    for (int kc = 0; kc < 4; ++kc)
        aq[kc] = frag_from_row(qsrc, kc, khalf);

    float mrow[8], lsum[8], acc0[8], acc1[8];
#pragma unroll
    for (int r = 0; r < 8; ++r) {
        mrow[r] = -INFINITY; lsum[r] = 0.f; acc0[r] = 0.f; acc1[r] = 0.f;
    }

    const _Float16* kbase  = K + (size_t)b * HW * CH;
    const float*    v0base = flow + (size_t)b * 2 * HW;
    const float*    v1base = v0base + HW;

    // prologue: fetch tile 0
    issue_ktile(kbase, 0, &ktile[0][0], tid);
    asm volatile("s_wait_asynccnt 0x0" ::: "memory");
    __syncthreads();

    for (int kt = 0; kt < QT16; ++kt) {
        int cur = kt & 1;
        if (kt + 1 < QT16)                                  // overlap copy of kt+1
            issue_ktile(kbase, kt + 1, &ktile[1 - cur][0], tid);

        // S tile = Q @ K^T ; B-fragments come from LDS (ds_load_b128)
        const _Float16* krow = &ktile[cur][m * CH];
        v8f s = {};
#pragma unroll
        for (int kc = 0; kc < 4; ++kc) {
            v16h bk = frag_from_row(krow, kc, khalf);
            s = __builtin_amdgcn_wmma_f32_16x16x32_f16(
                false, aq[kc], false, bk, (short)0, s, false, false);
        }

        float v0 = v0base[kt * 16 + m];
        float v1 = v1base[kt * 16 + m];

        // online softmax: row M=r spans the 16 lanes of this half-wave
#pragma unroll
        for (int r = 0; r < 8; ++r) {
            float sv = s[r] * scale;
            float rmax = sv;
            rmax = fmaxf(rmax, __shfl_xor(rmax, 1));
            rmax = fmaxf(rmax, __shfl_xor(rmax, 2));
            rmax = fmaxf(rmax, __shfl_xor(rmax, 4));
            rmax = fmaxf(rmax, __shfl_xor(rmax, 8));
            float mnew = fmaxf(mrow[r], rmax);
            float corr = __expf(mrow[r] - mnew);            // 0 on first tile (-inf)
            float p    = __expf(sv - mnew);
            mrow[r] = mnew;
            lsum[r] = lsum[r] * corr + p;                   // per-lane partials
            acc0[r] = acc0[r] * corr + p * v0;
            acc1[r] = acc1[r] * corr + p * v1;
        }

        asm volatile("s_wait_asynccnt 0x0" ::: "memory");   // tile kt+1 landed
        __syncthreads();                                    // publish to all waves
    }

    // final cross-lane reduction over the 16 key columns, normalize, store [b,2,h,w]
#pragma unroll
    for (int r = 0; r < 8; ++r) {
        float l = lsum[r], a0 = acc0[r], a1 = acc1[r];
#pragma unroll
        for (int mask = 1; mask < 16; mask <<= 1) {
            l  += __shfl_xor(l,  mask);
            a0 += __shfl_xor(a0, mask);
            a1 += __shfl_xor(a1, mask);
        }
        if (m == 0) {
            int row = row0 + r + khalf * 8;
            out[((size_t)b * 2 + 0) * HW + row] = a0 / l;
            out[((size_t)b * 2 + 1) * HW + row] = a1 / l;
        }
    }
}

extern "C" void kernel_launch(void* const* d_in, const int* in_sizes, int n_in,
                              void* d_out, int out_size, void* d_ws, size_t ws_size,
                              hipStream_t stream) {
    const float* feature0 = (const float*)d_in[0];
    const float* flow     = (const float*)d_in[1];
    const float* q_w      = (const float*)d_in[2];
    const float* q_b      = (const float*)d_in[3];
    const float* k_w      = (const float*)d_in[4];
    const float* k_b      = (const float*)d_in[5];
    float* out = (float*)d_out;

    _Float16* Qbuf = (_Float16*)d_ws;                       // 4*6144*128 f16
    _Float16* Kbuf = Qbuf + (size_t)BATCH * HW * CH;        // 4*6144*128 f16
    _Float16* qwh  = Kbuf + (size_t)BATCH * HW * CH;        // 128*128 f16
    _Float16* kwh  = qwh + CH * CH;                         // 128*128 f16

    cvt_weights_kernel<<<dim3((CH * CH) / 256), dim3(256), 0, stream>>>(q_w, k_w, qwh, kwh);
    proj_q_kernel<<<dim3(PROJ_BLOCKS), dim3(256), 0, stream>>>(feature0, qwh, q_b, Qbuf);
    proj_k_kernel<<<dim3(PROJ_BLOCKS), dim3(256), 0, stream>>>(Qbuf, kwh, k_b, Kbuf);
    attn_kernel  <<<dim3(ATTN_BLOCKS), dim3(512), 0, stream>>>(Qbuf, Kbuf, flow, out);
}
